// WeatherApdaptiveCNNBlock_37993280701146
// MI455X (gfx1250) — compile-verified, matrix operand order checked by reading, not run on hardware
//
#include <hip/hip_runtime.h>
#include <string.h>

typedef __attribute__((ext_vector_type(8)))  __bf16 v8bf;
typedef __attribute__((ext_vector_type(16))) __bf16 v16bf;
typedef __attribute__((ext_vector_type(8)))  float  v8f;

#define BATCH   16
#define CH      256
#define IMG     64          // H == W == 64
#define NPIX    4096        // 64*64
#define NEMB    8
#define XS_PAD  40          // LDS channel stride: 80B keeps 16B align, spreads banks

__device__ __forceinline__ unsigned short f2bf(float f) {
    union { float f; unsigned u; } v; v.f = f;
    unsigned r = v.u + 0x7FFFu + ((v.u >> 16) & 1u);   // round-to-nearest-even
    return (unsigned short)(r >> 16);
}

// CDNA5 async global->LDS copy (16B granule), tracked by ASYNCcnt
__device__ __forceinline__ void async_copy16(unsigned lds_off, const void* gaddr) {
    asm volatile("global_load_async_to_lds_b128 %0, %1, off"
                 :: "v"(lds_off), "v"(gaddr) : "memory");
}
__device__ __forceinline__ void wait_async0() {
    asm volatile("s_wait_asynccnt 0x0" ::: "memory");
}

// ---------------------------------------------------------------------------
// Kernel 1a: per-plane mean over HW (reads NCHW x, coalesced)
// ---------------------------------------------------------------------------
__global__ void __launch_bounds__(256) k_mean(const float* __restrict__ x,
                                              float* __restrict__ xmean) {
    int plane = blockIdx.x;                       // b*256 + c
    const float* p = x + (size_t)plane * NPIX;
    float s = 0.f;
    for (int i = threadIdx.x; i < NPIX; i += 256) s += p[i];
    __shared__ float red[256];
    red[threadIdx.x] = s;
    __syncthreads();
    for (int off = 128; off > 0; off >>= 1) {
        if (threadIdx.x < off) red[threadIdx.x] += red[threadIdx.x + off];
        __syncthreads();
    }
    if (threadIdx.x == 0) xmean[plane] = red[0] * (1.f / (float)NPIX);
}

// ---------------------------------------------------------------------------
// Kernel 1b: NCHW f32 -> NHWC bf16 transpose (one image row per block)
// xbf layout: [b][h][w][c], c contiguous -> conv staging is transpose-free
// ---------------------------------------------------------------------------
#define TP_PAD 264
__global__ void __launch_bounds__(256) k_nhwc(const float* __restrict__ x,
                                              unsigned short* __restrict__ xbf) {
    __shared__ unsigned short t[IMG * TP_PAD];    // [w][c]
    int h = blockIdx.x, b = blockIdx.y;
    const float* src = x + (size_t)b * CH * NPIX + (size_t)h * IMG;
    for (int idx = threadIdx.x; idx < CH * IMG; idx += 256) {
        int c = idx >> 6, w = idx & 63;           // coalesced 64-float runs per c
        t[w * TP_PAD + c] = f2bf(src[(size_t)c * NPIX + w]);
    }
    __syncthreads();
    unsigned short* dst = xbf + (((size_t)b * IMG + h) * IMG) * CH;
    for (int idx = threadIdx.x; idx < IMG * CH / 8; idx += 256) {
        int w = idx >> 5, cq = idx & 31;          // 32 x 16B chunks per pixel
        uint4 v = *reinterpret_cast<const uint4*>(&t[w * TP_PAD + cq * 8]);
        *reinterpret_cast<uint4*>(&dst[(size_t)w * CH + cq * 8]) = v;
    }
}

// ---------------------------------------------------------------------------
// Kernel 2: pooled = cls_conv_w @ xmean + b ; logits ; softmax -> mix (16x8)
// ---------------------------------------------------------------------------
__global__ void __launch_bounds__(256) k_classify(const float* __restrict__ xmean,
                                                   const float* __restrict__ cw,
                                                   const float* __restrict__ cb,
                                                   const float* __restrict__ dw,
                                                   const float* __restrict__ db,
                                                   float* __restrict__ mix) {
    __shared__ float pooled[BATCH][CH];
    __shared__ float logits[BATCH][NEMB];
    int d = threadIdx.x;
    for (int b = 0; b < BATCH; ++b) {
        float acc = cb[d];
        for (int c = 0; c < CH; ++c) acc += cw[d * CH + c] * xmean[b * CH + c];
        pooled[b][d] = acc;
    }
    __syncthreads();
    if (d < BATCH * NEMB) {
        int b = d >> 3, e = d & 7;
        float acc = db[e];
        for (int c = 0; c < CH; ++c) acc += dw[e * CH + c] * pooled[b][c];
        logits[b][e] = acc;
    }
    __syncthreads();
    if (d < BATCH) {
        int b = d;
        float m = -1e30f;
        for (int e = 0; e < NEMB; ++e) m = fmaxf(m, logits[b][e]);
        float ex[NEMB], s = 0.f;
        for (int e = 0; e < NEMB; ++e) { ex[e] = __expf(logits[b][e] - m); s += ex[e]; }
        float inv = 1.f / s;
        for (int e = 0; e < NEMB; ++e) mix[b * NEMB + e] = ex[e] * inv;
    }
}

// ---------------------------------------------------------------------------
// Kernel 3: W[b,o,i,kh,kw] = sum_e mix[b,e]*embed[e,o,i,kh,kw]  -> bf16
// wbuf layout: [((b*9 + khw)*256 + o)*256 + i]
// ---------------------------------------------------------------------------
__global__ void __launch_bounds__(256) k_synth(const float* __restrict__ embed,
                                                const float* __restrict__ mix,
                                                unsigned short* __restrict__ wbuf) {
    int b  = blockIdx.y;
    int oi = blockIdx.x * 256 + threadIdx.x;
    float m[NEMB];
#pragma unroll
    for (int e = 0; e < NEMB; ++e) m[e] = mix[b * NEMB + e];
    float acc[9];
#pragma unroll
    for (int k = 0; k < 9; ++k) acc[k] = 0.f;
    const float* src = embed + (size_t)oi * 9;
#pragma unroll
    for (int e = 0; e < NEMB; ++e) {
        const float* s2 = src + (size_t)e * (CH * CH * 9);
#pragma unroll
        for (int k = 0; k < 9; ++k) acc[k] += m[e] * s2[k];
    }
#pragma unroll
    for (int k = 0; k < 9; ++k)
        wbuf[((size_t)b * 9 + k) * (CH * CH) + oi] = f2bf(acc[k]);
}

// ---------------------------------------------------------------------------
// Kernel 4: implicit-GEMM 3x3 conv via v_wmma_f32_16x16x32_bf16
// per b: D[pix(M=4096), out(N=256)] = X^T(M,K=2304) * W(K,N)
// Block: 2 image rows (M=128) x 128 out-ch; 8 waves in 2(M) x 4(N) grid,
// each wave owns 64x32 = 4x2 WMMA accumulators (B fragments reused 4x).
// x tile staged with global_load_async_to_lds_b128 (no transpose: NHWC).
// grid: (32 row-pairs, 2 out-halves, 16 batch) x 256 threads
// ---------------------------------------------------------------------------
__device__ __forceinline__ v16bf pack16(v8bf lo, v8bf hi) {
    v16bf r;
#pragma unroll
    for (int j = 0; j < 8; ++j) { r[j] = lo[j]; r[j + 8] = hi[j]; }
    return r;
}

__global__ void __launch_bounds__(256) k_conv(const unsigned short* __restrict__ xbf,
                                              const unsigned short* __restrict__ wbuf,
                                              float* __restrict__ out) {
    // staged x tile: 4 rows (h0-1..h0+2) x 66 cols (w=-1..64) x 32 ch, bf16
    __shared__ unsigned short xs[4 * 66 * XS_PAD];

    const int h0    = blockIdx.x * 2;       // first of two image rows
    const int obase = blockIdx.y * 128;
    const int b     = blockIdx.z;

    const int tid   = threadIdx.x;
    const int lane  = tid & 31;
    const int wave  = tid >> 5;
    const int waveM = (wave & 1) * 64;      // 0 | 64  -> image row h0 | h0+1
    const int waveN = (wave >> 1) * 32;     // 0..96
    const int rimg  = waveM >> 6;           // 0 | 1
    const int ln    = lane & 15;
    const int koff  = (lane < 16) ? 0 : 8;  // 16-bit WMMA operand lane split

    v8f acc[4][2] = {};

    const unsigned short* xnhwc = xbf + (size_t)b * NPIX * CH;

    for (int ic = 0; ic < CH; ic += 32) {
        __syncthreads();    // previous tap loop done reading xs

        // stage 4 rows x 66 cols x 32 ch: 1056 chunks of 16B
        for (int q = tid; q < 4 * 66 * 4; q += 256) {
            int r    = q / 264;
            int rem  = q - r * 264;
            int c    = rem >> 2;            // 0..65  (w = c-1)
            int part = rem & 3;             // 16B sub-chunk (8 channels)
            unsigned lds_off =
                (unsigned)(((r * 66 + c) * XS_PAD + part * 8) * 2);
            unsigned lds_addr =
                (unsigned)(size_t)&xs[(r * 66 + c) * XS_PAD + part * 8];
            int hr = h0 + r - 1;
            if (c == 0 || c == 65 || hr < 0 || hr >= IMG) {
                *reinterpret_cast<uint4*>(&xs[(r * 66 + c) * XS_PAD + part * 8]) =
                    make_uint4(0u, 0u, 0u, 0u);
            } else {
                const unsigned short* g =
                    xnhwc + (((size_t)hr * IMG + (c - 1)) * CH + ic + part * 8);
                async_copy16(lds_addr, g);
            }
            (void)lds_off;
        }
        wait_async0();      // drain this wave's ASYNCcnt before the barrier
        __syncthreads();

        // 9 taps = 9 K-chunks of 32 channels
#pragma unroll
        for (int khw = 0; khw < 9; ++khw) {
            const int dh = khw / 3, dw = khw % 3;

            // A fragments (pixels x channels) from LDS: two aligned 16B reads
            v16bf A[4];
#pragma unroll
            for (int mt = 0; mt < 4; ++mt) {
                int col = mt * 16 + ln + dw;                  // 0..65
                const unsigned short* pa =
                    &xs[((rimg + dh) * 66 + col) * XS_PAD + koff];
                v8bf lo = *reinterpret_cast<const v8bf*>(pa);
                v8bf hi = *reinterpret_cast<const v8bf*>(pa + 16);
                A[mt] = pack16(lo, hi);
            }
            // B fragments (channels x out): straight from global (L2-resident)
            v16bf Bf[2];
#pragma unroll
            for (int nt = 0; nt < 2; ++nt) {
                const unsigned short* pb =
                    wbuf + (((size_t)b * 9 + khw) * CH + obase + waveN + nt * 16 + ln) * CH
                         + ic + koff;
                v8bf lo = *reinterpret_cast<const v8bf*>(pb);
                v8bf hi = *reinterpret_cast<const v8bf*>(pb + 16);
                Bf[nt] = pack16(lo, hi);
            }
#pragma unroll
            for (int mt = 0; mt < 4; ++mt)
#pragma unroll
                for (int nt = 0; nt < 2; ++nt)
                    acc[mt][nt] = __builtin_amdgcn_wmma_f32_16x16x32_bf16(
                        false, A[mt], false, Bf[nt], (short)0, acc[mt][nt], false, false);
        }
    }

    // epilogue: lane holds N=out-channel col, 8 consecutive M=pixel values
    const int h = h0 + rimg;
#pragma unroll
    for (int mt = 0; mt < 4; ++mt) {
#pragma unroll
        for (int nt = 0; nt < 2; ++nt) {
            int o  = obase + waveN + nt * 16 + ln;
            int w0 = mt * 16 + ((lane < 16) ? 0 : 8);
            float* p = out + ((size_t)(b * CH + o)) * NPIX + h * IMG + w0;
            v8f a = acc[mt][nt];
            *reinterpret_cast<float4*>(p)     = make_float4(a[0], a[1], a[2], a[3]);
            *reinterpret_cast<float4*>(p + 4) = make_float4(a[4], a[5], a[6], a[7]);
        }
    }
}

// ---------------------------------------------------------------------------
extern "C" void kernel_launch(void* const* d_in, const int* in_sizes, int n_in,
                              void* d_out, int out_size, void* d_ws, size_t ws_size,
                              hipStream_t stream) {
    const float* x     = (const float*)d_in[0];   // (16,256,64,64)
    const float* embed = (const float*)d_in[1];   // (8,256,256,3,3)
    const float* cw    = (const float*)d_in[2];   // (256,256)
    const float* cb    = (const float*)d_in[3];   // (256,)
    const float* dw    = (const float*)d_in[4];   // (8,256)
    const float* db    = (const float*)d_in[5];   // (8,)
    float* out = (float*)d_out;

    char* ws = (char*)d_ws;
    float*          xmean = (float*)ws;                               // 16 KB
    float*          mix   = (float*)(ws + 16384);                     // 512 B
    unsigned short* xbf   = (unsigned short*)(ws + 32768);            // 32 MiB NHWC
    unsigned short* wbuf  = (unsigned short*)(ws + 32768 +
                              (size_t)BATCH * CH * NPIX * 2);         // 18 MiB

    k_mean<<<BATCH * CH, 256, 0, stream>>>(x, xmean);
    k_nhwc<<<dim3(IMG, BATCH), 256, 0, stream>>>(x, xbf);
    k_classify<<<1, 256, 0, stream>>>(xmean, cw, cb, dw, db, mix);
    k_synth<<<dim3(CH * CH / 256, BATCH), 256, 0, stream>>>(embed, mix, wbuf);
    k_conv<<<dim3(IMG / 2, 2, BATCH), 256, 0, stream>>>(xbf, wbuf, out);
}
